// RPN_78735340470370
// MI455X (gfx1250) — matrix operand order repaired and checked by reading, not compiled, not run on hardware
//
#include <hip/hip_runtime.h>
#include <math.h>

// ---------- CDNA5 WMMA types ----------
typedef __bf16 v16bf __attribute__((ext_vector_type(16)));
typedef float  v8f   __attribute__((ext_vector_type(8)));

union Frag { v16bf v; uint4 u[2]; };

#define ANUM      3
#define N_POS     35464
#define N_ANC     106392      // N_POS * 3
#define PRE_NMS_N 6000
#define POST_NMS_N 1000
#define NMS_THR   0.7f
#define SORT_N    131072      // next pow2 >= N_ANC
#define KTOT      2304        // 9 taps * 256 in-channels
#define KCHUNK    128

// =====================================================================
// Kernel 1: repack weights into WMMA-GEMM layouts (fp32 -> bf16)
//   wmat[m][k] , k = (dy*3+dx)*256 + c   (512 x 2304)
//   hmat[o][c] , o 0..5 = score w, 6..17 = bbox w, 18..31 = 0 (32 x 512)
// =====================================================================
__global__ void prep_weights(const float* __restrict__ w1,
                             const float* __restrict__ wsc,
                             const float* __restrict__ wbb,
                             __bf16* __restrict__ wmat,
                             __bf16* __restrict__ hmat) {
    int idx = blockIdx.x * 256 + threadIdx.x;
    const int nw = 512 * KTOT;
    if (idx < nw) {
        int m = idx / KTOT, k = idx - m * KTOT;
        int tap = k >> 8, c = k & 255;
        int dy = tap / 3, dx = tap - 3 * dy;
        wmat[idx] = (__bf16)w1[((m * 256 + c) * 3 + dy) * 3 + dx];
    } else {
        int j = idx - nw;
        if (j < 32 * 512) {
            int o = j >> 9, c = j & 511;
            float v = 0.f;
            if (o < 6)       v = wsc[o * 512 + c];
            else if (o < 18) v = wbb[(o - 6) * 512 + c];
            hmat[j] = (__bf16)v;
        }
    }
}

// =====================================================================
// Kernel 2: fused 3x3 conv (implicit GEMM, bf16 WMMA) + bias + fused
// 1x1 heads (second WMMA GEMM) + softmax -> fg scores & bbox deltas.
// Workgroup = 8 waves (256 thr), tile = 512 M x 32 positions.
// =====================================================================
__launch_bounds__(256)
__global__ void rpn_conv_fused(const float* __restrict__ x, int H, int W, int pos_off,
                               const __bf16* __restrict__ wmat,
                               const __bf16* __restrict__ hmat,
                               const float* __restrict__ conv_b,
                               const float* __restrict__ score_b,
                               const float* __restrict__ bbox_b,
                               float* __restrict__ scores_out,
                               float* __restrict__ deltas_out) {
    __shared__ __bf16 patch[32][KCHUNK + 8];   // im2col tile (bf16), padded rows
    __shared__ __bf16 fbuf[32][512 + 8];       // hidden feature f (bf16), [pos][m]

    const int t    = threadIdx.x;
    const int lane = t & 31;
    const int wid  = t >> 5;
    const int half = lane >> 4;      // lane half: selects K sub-range in frags
    const int nn   = lane & 15;      // N column / M row inside a 16-tile
    const int HW   = H * W;
    const int ptile = blockIdx.x * 32;

    v8f acc[4][2] = {};              // 64 M (4 subtiles) x 32 N (2 subtiles)

    const int jj    = t & 31;        // position slot this thread fills
    const int kslot = t >> 5;        // 0..7

    for (int kc = 0; kc < KTOT; kc += KCHUNK) {
        __syncthreads();             // WAR: previous chunk consumed
        // ---- cooperative im2col fill (coalesced over positions) ----
        {
            int p  = ptile + jj;
            int hh = p / W, ww = p - hh * W;
            #pragma unroll
            for (int i = 0; i < KCHUNK / 8; i++) {
                int kk  = kslot + i * 8;
                int k   = kc + kk;
                int tap = k >> 8, c = k & 255;
                int dy  = tap / 3, dx = tap - 3 * dy;
                int ih  = hh + dy - 1, iw = ww + dx - 1;
                float v = 0.f;
                if (p < HW && (unsigned)ih < (unsigned)H && (unsigned)iw < (unsigned)W)
                    v = x[(c * H + ih) * W + iw];
                patch[jj][kk] = (__bf16)v;
            }
        }
        __syncthreads();
        // ---- 4 k-steps of 32 over this chunk ----
        #pragma unroll
        for (int ks = 0; ks < KCHUNK; ks += 32) {
            Frag b[2];
            #pragma unroll
            for (int ns = 0; ns < 2; ns++) {
                const __bf16* bp = &patch[ns * 16 + nn][ks + half * 8];
                b[ns].u[0] = *(const uint4*)bp;        // K = h*8 + 0..7
                b[ns].u[1] = *(const uint4*)(bp + 16); // K = 16 + h*8 + 0..7
            }
            #pragma unroll
            for (int ms = 0; ms < 4; ms++) {
                int m = wid * 64 + ms * 16 + nn;
                const __bf16* ap = wmat + (size_t)m * KTOT + (kc + ks) + half * 8;
                Frag a;
                a.u[0] = *(const uint4*)ap;            // K = h*8 + 0..7
                a.u[1] = *(const uint4*)(ap + 16);     // K = 16 + h*8 + 0..7
                __builtin_prefetch((const void*)(ap + KCHUNK), 0, 3);
                #pragma unroll
                for (int ns = 0; ns < 2; ns++)
                    acc[ms][ns] = __builtin_amdgcn_wmma_f32_16x16x32_bf16(
                        false, a.v, false, b[ns].v, (short)0, acc[ms][ns], false, false);
            }
        }
    }

    // ---- bias + stash f (bf16) into LDS [pos][m] for the head GEMM ----
    __syncthreads();
    #pragma unroll
    for (int ms = 0; ms < 4; ms++)
        #pragma unroll
        for (int ns = 0; ns < 2; ns++)
            #pragma unroll
            for (int v = 0; v < 8; v++) {
                int m = wid * 64 + ms * 16 + v + 8 * half;   // C/D layout: M=v+8h
                fbuf[ns * 16 + nn][m] = (__bf16)(acc[ms][ns][v] + conv_b[m]);
            }
    __syncthreads();

    // ---- head GEMM (32x512) x (512x32) on wave 0, then emit outputs ----
    if (wid == 0) {
        v8f hacc[2][2] = {};
        for (int ks = 0; ks < 512; ks += 32) {
            Frag b[2];
            #pragma unroll
            for (int ns = 0; ns < 2; ns++) {
                const __bf16* bp = &fbuf[ns * 16 + nn][ks + half * 8];
                b[ns].u[0] = *(const uint4*)bp;
                b[ns].u[1] = *(const uint4*)(bp + 16);
            }
            #pragma unroll
            for (int ms = 0; ms < 2; ms++) {
                const __bf16* ap = hmat + (ms * 16 + nn) * 512 + ks + half * 8;
                Frag a;
                a.u[0] = *(const uint4*)ap;
                a.u[1] = *(const uint4*)(ap + 16);
                #pragma unroll
                for (int ns = 0; ns < 2; ns++)
                    hacc[ms][ns] = __builtin_amdgcn_wmma_f32_16x16x32_bf16(
                        false, a.v, false, b[ns].v, (short)0, hacc[ms][ns], false, false);
            }
        }
        #pragma unroll
        for (int ns = 0; ns < 2; ns++) {
            int p = ptile + ns * 16 + nn;
            if (p >= HW) continue;
            size_t pg = (size_t)(pos_off + p);
            #pragma unroll
            for (int ms = 0; ms < 2; ms++)
                #pragma unroll
                for (int v = 0; v < 8; v++) {
                    int o = ms * 16 + v + 8 * half;
                    if (o >= 6 && o < 18)
                        deltas_out[pg * 12 + (o - 6)] = hacc[ms][ns][v] + bbox_b[o - 6];
                }
            if (half == 0) {
                // this lane holds cls channels 0..7 in hacc[0][ns][0..7]
                #pragma unroll
                for (int a = 0; a < ANUM; a++) {
                    float s0 = hacc[0][ns][a]        + score_b[a];
                    float s1 = hacc[0][ns][a + ANUM] + score_b[a + ANUM];
                    // softmax over 2 classes -> fg prob = sigmoid(s1 - s0)
                    scores_out[pg * 3 + a] = 1.f / (1.f + expf(s0 - s1));
                }
            }
        }
    }
}

// =====================================================================
// Kernel 3: anchors + bbox_transform_inv + clip + sort-key build
// =====================================================================
__global__ void make_proposals(const float* __restrict__ scores,
                               const float* __restrict__ deltas,
                               const float* __restrict__ im_info,
                               float* __restrict__ proposals,
                               unsigned long long* __restrict__ keys) {
    int tid = blockIdx.x * 256 + threadIdx.x;
    if (tid >= SORT_N) return;
    if (tid >= N_ANC) { keys[tid] = 0xFFFFFFFFFFFFFFFFull; return; }   // pad: sorts last

    int pos = tid / 3, a = tid - pos * 3;
    const int offs[6] = {0, 26624, 33280, 34944, 35360, 35464};
    const int Wl[5]   = {208, 104, 52, 26, 13};
    const int Sl[5]   = {4, 8, 16, 32, 64};
    int lev = 0;
    while (pos >= offs[lev + 1]) lev++;
    int lp = pos - offs[lev];
    int Wn = Wl[lev];
    float fs = (float)Sl[lev];
    int gx = lp % Wn, gy = lp / Wn;

    float ratio = (a == 0) ? 0.5f : (a == 1 ? 1.f : 2.f);
    float ws0 = roundf(sqrtf(fs * fs / ratio));
    float hs0 = roundf(ws0 * ratio);
    float wsz = ws0 * 8.f, hsz = hs0 * 8.f;
    float c  = (fs - 1.f) * 0.5f;
    float sx = gx * fs, sy = gy * fs;
    float ax1 = c - 0.5f * (wsz - 1.f) + sx;
    float ay1 = c - 0.5f * (hsz - 1.f) + sy;
    float ax2 = c + 0.5f * (wsz - 1.f) + sx;
    float ay2 = c + 0.5f * (hsz - 1.f) + sy;

    float w  = ax2 - ax1 + 1.f, h = ay2 - ay1 + 1.f;
    float cx = ax1 + 0.5f * w,  cy = ay1 + 0.5f * h;
    const float* d = deltas + (size_t)pos * 12 + a * 4;
    float pcx = d[0] * w + cx, pcy = d[1] * h + cy;
    float pw  = expf(d[2]) * w, ph = expf(d[3]) * h;

    float imH = im_info[0], imW = im_info[1];
    float x1 = fminf(fmaxf(pcx - 0.5f * pw, 0.f), imW - 1.f);
    float y1 = fminf(fmaxf(pcy - 0.5f * ph, 0.f), imH - 1.f);
    float x2 = fminf(fmaxf(pcx + 0.5f * pw, 0.f), imW - 1.f);
    float y2 = fminf(fmaxf(pcy + 0.5f * ph, 0.f), imH - 1.f);
    proposals[tid * 4 + 0] = x1;
    proposals[tid * 4 + 1] = y1;
    proposals[tid * 4 + 2] = x2;
    proposals[tid * 4 + 3] = y2;

    // key: descending score, ascending index (stable like lax.top_k)
    unsigned ub = __float_as_uint(scores[tid]);
    ub = (ub & 0x80000000u) ? ~ub : (ub | 0x80000000u);
    keys[tid] = ((unsigned long long)(~ub) << 32) | (unsigned)tid;
}

// =====================================================================
// Kernel 4: global bitonic sort step (ascending composite keys)
// =====================================================================
__global__ void sort_step(unsigned long long* __restrict__ keys, int k, int j) {
    int i = blockIdx.x * 256 + threadIdx.x;
    int ixj = i ^ j;
    if (ixj > i && ixj < SORT_N) {
        unsigned long long a = keys[i], b = keys[ixj];
        bool up = ((i & k) == 0);
        if ((a > b) == up) { keys[i] = b; keys[ixj] = a; }
    }
}

// =====================================================================
// Kernel 5: gather top PRE_NMS boxes/scores in sorted order
// =====================================================================
__global__ void gather_top(const unsigned long long* __restrict__ keys,
                           const float* __restrict__ proposals,
                           const float* __restrict__ scores,
                           float* __restrict__ tboxes, float* __restrict__ tscores) {
    int r = blockIdx.x * 256 + threadIdx.x;
    if (r >= PRE_NMS_N) return;
    unsigned idx = (unsigned)(keys[r] & 0xFFFFFFFFu);
    if (idx >= N_ANC) idx = 0;
    tboxes[r * 4 + 0] = proposals[idx * 4 + 0];
    tboxes[r * 4 + 1] = proposals[idx * 4 + 1];
    tboxes[r * 4 + 2] = proposals[idx * 4 + 2];
    tboxes[r * 4 + 3] = proposals[idx * 4 + 3];
    tscores[r] = scores[idx];
}

// =====================================================================
// Kernel 6: sequential NMS (single workgroup, scores in LDS) + output
// =====================================================================
__launch_bounds__(1024)
__global__ void nms_kernel(const float* __restrict__ boxes,
                           const float* __restrict__ tscores,
                           float* __restrict__ out) {
    __shared__ float s[PRE_NMS_N];
    __shared__ float rv[32];
    __shared__ int   ri[32];
    __shared__ int   keepIdx[POST_NMS_N];
    __shared__ int   keepValid[POST_NMS_N];
    __shared__ float selbox[4];

    const int t = threadIdx.x, lane = t & 31, wid = t >> 5;
    for (int e = t; e < PRE_NMS_N; e += 1024) s[e] = tscores[e];
    __syncthreads();

    for (int it = 0; it < POST_NMS_N; it++) {
        // ---- argmax (first occurrence on ties, like jnp.argmax) ----
        float bv = -INFINITY; int bi = 0;
        for (int e = t; e < PRE_NMS_N; e += 1024) {
            float v = s[e];
            if (v > bv) { bv = v; bi = e; }
        }
        for (int off = 16; off > 0; off >>= 1) {
            float ov = __shfl_down(bv, off);
            int   oi = __shfl_down(bi, off);
            if (ov > bv || (ov == bv && oi < bi)) { bv = ov; bi = oi; }
        }
        if (lane == 0) { rv[wid] = bv; ri[wid] = bi; }
        __syncthreads();
        if (wid == 0) {
            bv = rv[lane]; bi = ri[lane];
            for (int off = 16; off > 0; off >>= 1) {
                float ov = __shfl_down(bv, off);
                int   oi = __shfl_down(bi, off);
                if (ov > bv || (ov == bv && oi < bi)) { bv = ov; bi = oi; }
            }
            if (lane == 0) {
                keepIdx[it]   = bi;
                keepValid[it] = (bv > -INFINITY) ? 1 : 0;
                selbox[0] = boxes[bi * 4 + 0];
                selbox[1] = boxes[bi * 4 + 1];
                selbox[2] = boxes[bi * 4 + 2];
                selbox[3] = boxes[bi * 4 + 3];
                s[bi] = -INFINITY;
            }
        }
        __syncthreads();
        // ---- suppress IoU > thr ----
        float bx1 = selbox[0], by1 = selbox[1], bx2 = selbox[2], by2 = selbox[3];
        float areab = (bx2 - bx1 + 1.f) * (by2 - by1 + 1.f);
        for (int e = t; e < PRE_NMS_N; e += 1024) {
            if (s[e] == -INFINITY) continue;
            float x1 = boxes[e * 4 + 0], y1 = boxes[e * 4 + 1];
            float x2 = boxes[e * 4 + 2], y2 = boxes[e * 4 + 3];
            float area = (x2 - x1 + 1.f) * (y2 - y1 + 1.f);
            float xx1 = fmaxf(bx1, x1), yy1 = fmaxf(by1, y1);
            float xx2 = fminf(bx2, x2), yy2 = fminf(by2, y2);
            float iw = fmaxf(0.f, xx2 - xx1 + 1.f);
            float ih = fmaxf(0.f, yy2 - yy1 + 1.f);
            float inter = iw * ih;
            if (inter / (area + areab - inter) > NMS_THR) s[e] = -INFINITY;
        }
        __syncthreads();
    }

    // ---- final (1000 x 6) output: [0, x1, y1, x2, y2, score] * valid ----
    for (int r = t; r < POST_NMS_N; r += 1024) {
        int i = keepIdx[r];
        float v = keepValid[r] ? 1.f : 0.f;
        out[r * 6 + 0] = 0.f;
        out[r * 6 + 1] = boxes[i * 4 + 0] * v;
        out[r * 6 + 2] = boxes[i * 4 + 1] * v;
        out[r * 6 + 3] = boxes[i * 4 + 2] * v;
        out[r * 6 + 4] = boxes[i * 4 + 3] * v;
        out[r * 6 + 5] = tscores[i] * v;
    }
}

// =====================================================================
extern "C" void kernel_launch(void* const* d_in, const int* in_sizes, int n_in,
                              void* d_out, int out_size, void* d_ws, size_t ws_size,
                              hipStream_t stream) {
    (void)in_sizes; (void)n_in; (void)out_size; (void)ws_size;
    const float* feats[5] = {(const float*)d_in[0], (const float*)d_in[1],
                             (const float*)d_in[2], (const float*)d_in[3],
                             (const float*)d_in[4]};
    const float* im_info = (const float*)d_in[5];
    const float* w1  = (const float*)d_in[6];
    const float* b1  = (const float*)d_in[7];
    const float* wsc = (const float*)d_in[8];
    const float* bsc = (const float*)d_in[9];
    const float* wbb = (const float*)d_in[10];
    const float* bbb = (const float*)d_in[11];

    char* ws = (char*)d_ws;
    size_t off = 0;
    auto carve = [&](size_t bytes) -> char* {
        char* p = ws + off;
        off = (off + bytes + 255) & ~(size_t)255;
        return p;
    };
    __bf16* wmat  = (__bf16*)carve((size_t)512 * KTOT * 2);
    __bf16* hmat  = (__bf16*)carve((size_t)32 * 512 * 2);
    float* scores = (float*)carve((size_t)N_ANC * 4);
    float* deltas = (float*)carve((size_t)N_POS * 12 * 4);
    float* props  = (float*)carve((size_t)N_ANC * 16);
    unsigned long long* keys = (unsigned long long*)carve((size_t)SORT_N * 8);
    float* tboxes  = (float*)carve((size_t)PRE_NMS_N * 16);
    float* tscores = (float*)carve((size_t)PRE_NMS_N * 4);

    const int prep_n = 512 * KTOT + 32 * 512;
    prep_weights<<<(prep_n + 255) / 256, 256, 0, stream>>>(w1, wsc, wbb, wmat, hmat);

    const int Hs[5]   = {128, 64, 32, 16, 8};
    const int Wsv[5]  = {208, 104, 52, 26, 13};
    const int poff[5] = {0, 26624, 33280, 34944, 35360};
    for (int l = 0; l < 5; l++) {
        int HW = Hs[l] * Wsv[l];
        int tiles = (HW + 31) / 32;
        rpn_conv_fused<<<tiles, 256, 0, stream>>>(feats[l], Hs[l], Wsv[l], poff[l],
                                                  wmat, hmat, b1, bsc, bbb,
                                                  scores, deltas);
    }

    make_proposals<<<SORT_N / 256, 256, 0, stream>>>(scores, deltas, im_info, props, keys);

    for (int k = 2; k <= SORT_N; k <<= 1)
        for (int j = k >> 1; j > 0; j >>= 1)
            sort_step<<<SORT_N / 256, 256, 0, stream>>>(keys, k, j);

    gather_top<<<(PRE_NMS_N + 255) / 256, 256, 0, stream>>>(keys, props, scores,
                                                            tboxes, tscores);

    nms_kernel<<<1, 1024, 0, stream>>>(tboxes, tscores, (float*)d_out);
}
